// EngramMHFFNO1D_38199439131351
// MI455X (gfx1250) — matrix operand order repaired and checked by reading, not compiled, not run on hardware
//
#include <hip/hip_runtime.h>
#include <math.h>

typedef __attribute__((ext_vector_type(16))) _Float16 v16h;
typedef __attribute__((ext_vector_type(8)))  float    v8f;
typedef __attribute__((ext_vector_type(2)))  float    v2f;
typedef __attribute__((ext_vector_type(4)))  unsigned int eng_u32x4;
typedef __attribute__((ext_vector_type(8)))  int          eng_i32x8;
typedef __attribute__((ext_vector_type(4)))  int          eng_i32x4;

#define S_LEN   16384
#define BATCH   16
#define HID     64
#define MODES   16
#define NP_SP   500
#define NP_FR   200
#define ED      8
#define LAYERS  4
#define KSPLIT  8            /* forward-DFT K split (parallelism) */
#define SSPLIT  8            /* fused kernel S split */
#define TWO_PI_OVER_S 3.83495196971410293e-4f  /* 2*pi/16384 */

#ifndef __has_builtin
#define __has_builtin(x) 0
#endif
#if __has_builtin(__builtin_amdgcn_wmma_f32_16x16x4_f32)
#define HAVE_WMMA_F32 1
#else
#define HAVE_WMMA_F32 0
#endif
#if __has_builtin(__builtin_amdgcn_tensor_load_to_lds)
#define HAVE_TDM 1
#else
#define HAVE_TDM 0
#endif

/* Device-pass diagnostics so the compile log confirms which CDNA5 paths are live. */
#if defined(__AMDGCN__)
#if HAVE_TDM
#pragma message("ENG gfx1250 device pass: TDM tensor_load_to_lds path ENABLED")
#else
#pragma message("ENG gfx1250 device pass: TDM path DISABLED (builtin missing) - manual staging")
#endif
#if HAVE_WMMA_F32
#pragma message("ENG gfx1250 device pass: f32 WMMA 16x16x4 path ENABLED")
#else
#pragma message("ENG gfx1250 device pass: f32 WMMA path DISABLED - scalar fallback")
#endif
#endif

// ---------------------------------------------------------------------------
// TDM: DMA a 16-row x 128-dword f32 tile (row stride S_LEN) into LDS with
// 4-dword padding every 128 dwords (-> LDS row stride 132). D# per ISA 8.3-8.5.
// ---------------------------------------------------------------------------
#if HAVE_TDM
__device__ __forceinline__ void eng_tdm_tile16x128(unsigned ldsOff, const float* gsrc) {
  unsigned long long ga = (unsigned long long)(size_t)gsrc;
  unsigned galo = (unsigned)__builtin_amdgcn_readfirstlane((int)(unsigned)(ga & 0xffffffffULL));
  unsigned gahi = (unsigned)__builtin_amdgcn_readfirstlane((int)(unsigned)(ga >> 32));
  unsigned lo   = (unsigned)__builtin_amdgcn_readfirstlane((int)ldsOff);
  eng_u32x4 g0;
  g0[0] = 1u;                                   // count=1 (valid user descriptor)
  g0[1] = lo;                                   // lds_addr (bytes)
  g0[2] = galo;                                 // global_addr[31:0]
  g0[3] = (gahi & 0x1ffffffu) | (2u << 30);     // global_addr[56:32] | type=2
  eng_i32x8 g1;
  g1[0] = (int)((2u << 16) | (1u << 20) | (6u << 22) | (3u << 25));
          // data_size=4B, pad_enable, pad_interval=128dw, pad_amount=4dw
  g1[1] = (int)(16384u << 16);                  // tensor_dim0 = 16384 (low16 @ [63:48])
  g1[2] = (int)(16u << 16);                     // dim0 hi=0 | tensor_dim1 = 16
  g1[3] = (int)(128u << 16);                    // dim1 hi=0 | tile_dim0 = 128
  g1[4] = 16;                                   // tile_dim1 = 16, tile_dim2 = 0
  g1[5] = 16384;                                // tensor_dim0_stride[31:0]
  g1[6] = 0;                                    // stride hi | tensor_dim1_stride lo
  g1[7] = 0;
  eng_i32x4 gz = {0, 0, 0, 0};
#if __clang_major__ >= 23
  eng_i32x8 gz8 = {};
  __builtin_amdgcn_tensor_load_to_lds(g0, g1, gz, gz, gz8, 0);
#else
  __builtin_amdgcn_tensor_load_to_lds(g0, g1, gz, gz, 0);
#endif
}
#endif

// ---------------------------------------------------------------------------
// Basis init. Phase reduced mod S (S = 2^14) so cosf/sinf args stay accurate.
// invB: inverse basis in f16 B-operand lane layout, contiguous k = 16*(lane/16) + j (32x16 per s-tile).
// ---------------------------------------------------------------------------
__global__ void eng_init_invB(_Float16* __restrict__ invB) {
  int gid = blockIdx.x * 256 + threadIdx.x;        // 1024 tiles * 512
  int t = gid >> 9, r = gid & 511, L = r >> 4, j = r & 15;
  int s = t * 16 + (L & 15);
  int K = ((L < 16) ? 0 : 16) + j;
  int mode = (K < 16) ? K : (K - 16);
  int ph = (mode * s) & (S_LEN - 1);
  float ang = TWO_PI_OVER_S * (float)ph;
  invB[gid] = (_Float16)((K < 16) ? cosf(ang) : sinf(ang));
}

// fwdB: forward basis in exact WMMA f32 B-operand order (4x16 per K-step, 2 N-tiles).
__global__ void eng_init_fwdB(float* __restrict__ fwdB) {
  int gid = blockIdx.x * 256 + threadIdx.x;        // 4096 k4 * 2 nt * 64
  int k4 = gid >> 7, r = gid & 127, nt = r >> 6, rr = r & 63;
  int L = rr >> 1, hsel = rr & 1;
  int n = nt * 16 + (L & 15);
  int s = k4 * 4 + ((L < 16) ? 0 : 2) + hsel;
  int mode = (n < 16) ? n : (n - 16);
  int ph = (mode * s) & (S_LEN - 1);
  float ang = TWO_PI_OVER_S * (float)ph;
  fwdB[gid] = (n < 16) ? cosf(ang) : -sinf(ang);
}

// Lifting: h[b,d,s] = sum_c x[b,c,s]*W[c,d] + bias[d]
__global__ void eng_lift(const float* __restrict__ x, const float* __restrict__ W,
                         const float* __restrict__ bias, float* __restrict__ h) {
  int bd = blockIdx.x;
  int b = bd >> 6, d = bd & 63;
  int s = blockIdx.y * 256 + threadIdx.x;
  float acc = bias[d];
  for (int c = 0; c < 3; ++c)
    acc += x[((size_t)(b * 3 + c)) * S_LEN + s] * W[c * HID + d];
  h[((size_t)bd) * S_LEN + s] = acc;
}

// ---------------------------------------------------------------------------
// Forward 16-mode DFT, K split 8 ways: modp[split,b,c,n] partials.
// Per chunk: TDM (or float4) stage 16x128 tile -> 16x v_wmma_f32_16x16x4_f32
// -> also emit the per-ct channel-sum partials (saves a full h re-read).
// fwdB fragment addressing strength-reduced to one pointer + immediate offsets.
// Deterministic fixed-order cross-wave reduction; no atomics.
// ---------------------------------------------------------------------------
__global__ void eng_fwd_dft(const float* __restrict__ h, const float* __restrict__ fwdB,
                            float* __restrict__ modp, float* __restrict__ sCpart) {
  __shared__ alignas(16) float stage[2][16 * 132];
  __shared__ float red[128 * 16];
  int b = blockIdx.x >> 2, ct = blockIdx.x & 3, split = blockIdx.y;
  int tid = threadIdx.x, wv = tid >> 5, lane = tid & 31;
  const float* hbase = h + ((size_t)(b * HID + ct * 16)) * S_LEN;
  const int k0 = split * (S_LEN / KSPLIT);
  v8f acc0 = {}, acc1 = {};
  int row = lane & 15;
  // per-wave fwdB fragment base: element (k4,nt,lane) at k4*128 + nt*64 + lane*2
  const float* fBw = fwdB + (size_t)(k0 >> 2) * 128 + (size_t)wv * (8 * 128) + lane * 2;

#if HAVE_TDM
  if (wv == 0) eng_tdm_tile16x128((unsigned)(size_t)&stage[0][0], hbase + k0);
#endif
  for (int ci = 0; ci < 16; ++ci) {
    int kc = k0 + ci * 128;
#if HAVE_TDM
    if (wv == 0) {
      if (ci + 1 < 16) {
        eng_tdm_tile16x128((unsigned)(size_t)&stage[(ci + 1) & 1][0], hbase + kc + 128);
        __builtin_amdgcn_s_wait_tensorcnt(1);   // oldest (current chunk) complete
      } else {
        __builtin_amdgcn_s_wait_tensorcnt(0);
      }
    }
    __syncthreads();
    const float* st = stage[ci & 1];
#else
    __syncthreads();
    for (int q = tid; q < 512; q += 128) {      // 16 rows x 32 float4
      int m = q >> 5, c4 = (q & 31) * 4;
      float4 v = *reinterpret_cast<const float4*>(hbase + (size_t)m * S_LEN + kc + c4);
      *reinterpret_cast<float4*>(&stage[0][m * 132 + c4]) = v;
    }
    __syncthreads();
    const float* st = stage[0];
#endif
    for (int step = 0; step < 8; ++step) {
      int kbase = wv * 32 + step * 4;
      v2f a;
      a.x = st[row * 132 + kbase + ((lane < 16) ? 0 : 2)];
      a.y = st[row * 132 + kbase + ((lane < 16) ? 1 : 3)];
#if HAVE_WMMA_F32
      v2f b0 = *reinterpret_cast<const v2f*>(fBw + step * 128);
      v2f b1 = *reinterpret_cast<const v2f*>(fBw + step * 128 + 64);
      acc0 = __builtin_amdgcn_wmma_f32_16x16x4_f32(false, a, false, b0, (short)0, acc0, false, false);
      acc1 = __builtin_amdgcn_wmma_f32_16x16x4_f32(false, a, false, b1, (short)0, acc1, false, false);
#else
      int ncol = lane & 15;
      for (int v = 0; v < 8; ++v) {
        int m = v + ((lane < 16) ? 0 : 8);
        float s0 = 0.f, s1 = 0.f;
        for (int kk = 0; kk < 4; ++kk) {
          float hv = st[m * 132 + kbase + kk];
          int sIdx = kc + kbase + kk;
          int ph = (ncol * sIdx) & (S_LEN - 1);
          float ang = TWO_PI_OVER_S * (float)ph;
          s0 += hv * cosf(ang);
          s1 -= hv * sinf(ang);
        }
        acc0[v] += s0; acc1[v] += s1;
      }
#endif
    }
    fBw += 32 * 128;                            // advance one chunk (32 k4 steps)
    // channel-sum partial for the spatial engram (free: tile already in LDS)
    float ps = 0.f;
    for (int m = 0; m < 16; ++m) ps += st[m * 132 + tid];
    sCpart[((size_t)(ct * BATCH + b)) * S_LEN + kc + tid] = ps;
#if HAVE_TDM
    __syncthreads();                            // buffer-reuse fence for DMA
#endif
  }
  __syncthreads();
  for (int j = 0; j < 8; ++j) { red[tid * 16 + j] = acc0[j]; red[tid * 16 + 8 + j] = acc1[j]; }
  __syncthreads();
  if (tid < 32) {
    for (int j = 0; j < 16; ++j) {
      float ssum = 0.f;
      for (int w = 0; w < 4; ++w) ssum += red[(w * 32 + tid) * 16 + j];
      int nt = j >> 3, v = j & 7;
      int c = ct * 16 + v + ((tid < 16) ? 0 : 8);
      int n = nt * 16 + (tid & 15);
      modp[((size_t)((split * BATCH + b) * HID + c)) * 32 + n] = ssum;
    }
  }
}

// Reduce K-split partials (fixed order).
__global__ void eng_modes_red(const float* __restrict__ modp, float* __restrict__ modes) {
  int gid = blockIdx.x * 256 + threadIdx.x;      // 16*64*32
  float s = 0.f;
  for (int sp = 0; sp < KSPLIT; ++sp) s += modp[(size_t)sp * (BATCH * HID * 32) + gid];
  modes[gid] = s;
}

// Combine per-ct channel-sum partials into sC[b,s].
__global__ void eng_scomb(const float* __restrict__ sCpart, float* __restrict__ sC) {
  int b = blockIdx.x;
  int s = blockIdx.y * 256 + threadIdx.x;
  float acc = 0.f;
  for (int ct = 0; ct < 4; ++ct) acc += sCpart[((size_t)(ct * BATCH + b)) * S_LEN + s];
  sC[(size_t)b * S_LEN + s] = acc;
}

// Spatial engram: window hash -> idx[b,s] + per-chunk embedding-sum partials.
__global__ void eng_spatial(const float* __restrict__ sC, const float* __restrict__ emb,
                            int* __restrict__ idxb, float* __restrict__ embpart) {
  __shared__ float red[256 * ED];
  int b = blockIdx.x, chunk = blockIdx.y, tid = threadIdx.x;
  const float* sb = sC + (size_t)b * S_LEN;
  const int base = chunk * (S_LEN / SSPLIT);
  float acc[ED];
  for (int e = 0; e < ED; ++e) acc[e] = 0.f;
  for (int s = base + tid; s < base + S_LEN / SSPLIT; s += 256) {
    int p0 = (s - 2 < 0) ? 0 : s - 2;
    int p1 = (s - 1 < 0) ? 0 : s - 1;
    int p3 = (s + 1 > S_LEN - 1) ? S_LEN - 1 : s + 1;
    float w = sb[p0] + sb[p1] + sb[s] + sb[p3];
    int iv = (int)(w * 31.0f);                  // trunc toward zero
    iv %= NP_SP; if (iv < 0) iv += NP_SP;
    idxb[(size_t)b * S_LEN + s] = iv;
    const float* e8 = emb + (size_t)iv * ED;
    for (int e = 0; e < ED; ++e) acc[e] += e8[e];
  }
  for (int e = 0; e < ED; ++e) red[tid * ED + e] = acc[e];
  __syncthreads();
  for (int off = 128; off > 0; off >>= 1) {
    if (tid < off)
      for (int e = 0; e < ED; ++e) red[tid * ED + e] += red[(tid + off) * ED + e];
    __syncthreads();
  }
  if (tid < ED) embpart[(b * SSPLIT + chunk) * ED + tid] = red[tid];
}

// Multi-head complex channel mixing in mode space (tiny).
__global__ void eng_mhf(const float* __restrict__ modes, const float* __restrict__ Wr,
                        const float* __restrict__ Wi, float* __restrict__ ofbuf) {
  int gid = blockIdx.x * 256 + threadIdx.x;     // 16*4*16*16
  int m = gid & 15, o = (gid >> 4) & 15, hh = (gid >> 8) & 3, b = gid >> 10;
  float fr = 0.f, fi = 0.f;
  for (int i = 0; i < 16; ++i) {
    const float* mp = modes + ((size_t)(b * HID + hh * 16 + i)) * 32;
    float xr = mp[m], xi = mp[16 + m];
    float wr = Wr[(((size_t)hh * 16 + i) * 16 + o) * 16 + m];
    float wi = Wi[(((size_t)hh * 16 + i) * 16 + o) * 16 + m];
    fr += xr * wr - xi * wi;
    fi += xr * wi + xi * wr;
  }
  int c = hh * 16 + o;
  ofbuf[((size_t)(b * HID + c)) * 32 + m] = fr;
  ofbuf[((size_t)(b * HID + c)) * 32 + 16 + m] = fi;
}

// Spectral engram: magnitudes -> hash -> embedding -> linear -> frproj[b,c,m]
__global__ void eng_freng(const float* __restrict__ modes, const float* __restrict__ fremb,
                          const float* __restrict__ frW, const float* __restrict__ frb,
                          float* __restrict__ frproj) {
  __shared__ float mag[MODES];
  __shared__ float e8[ED];
  __shared__ int sidx;
  int b = blockIdx.x, tid = threadIdx.x;
  if (tid < MODES) {
    float s = 0.f;
    for (int c = 0; c < HID; ++c) {
      const float* mp = modes + ((size_t)(b * HID + c)) * 32;
      float xr = mp[tid], xi = mp[16 + tid];
      s += sqrtf(xr * xr + xi * xi);
    }
    mag[tid] = s * (1.0f / 64.0f);
  }
  __syncthreads();
  if (tid == 0) {
    int acc = 0;
    for (int m = 0; m < MODES; ++m) acc += (int)(mag[m] * 1000.0f);
    int iv = acc % NP_FR; if (iv < 0) iv += NP_FR;
    sidx = iv;
  }
  __syncthreads();
  if (tid < ED) e8[tid] = fremb[(size_t)sidx * ED + tid];
  __syncthreads();
  for (int o = tid; o < HID * MODES; o += 256) {
    float v = frb[o];
    for (int e = 0; e < ED; ++e) v += e8[e] * frW[e * (HID * MODES) + o];
    int c = o >> 4, m = o & 15;
    frproj[((size_t)(b * HID + c)) * MODES + m] = v;
  }
}

// Gating MLP. Branch means via DC-mode identities (no branch materialization).
__global__ void eng_gate(const float* __restrict__ embpart, const float* __restrict__ ofbuf,
                         const float* __restrict__ frproj, const float* __restrict__ spW,
                         const float* __restrict__ spb, const float* __restrict__ W1,
                         const float* __restrict__ b1, const float* __restrict__ W2,
                         const float* __restrict__ b2, float* __restrict__ gatew) {
  __shared__ float g[3 * HID];
  __shared__ float h1[64];
  __shared__ float lg[3];
  int b = blockIdx.x, j = threadIdx.x;          // 64 threads
  const float invS = 1.0f / (float)S_LEN;
  float sv = spb[j];
  for (int e = 0; e < ED; ++e) {
    float es = 0.f;
    for (int ch = 0; ch < SSPLIT; ++ch) es += embpart[(b * SSPLIT + ch) * ED + e];
    sv += (es * invS) * spW[e * HID + j];
  }
  g[j] = sv;
  g[HID + j]     = ofbuf[((size_t)(b * HID + j)) * 32] * invS;      // mh mean = Re(DC)/S
  g[2 * HID + j] = frproj[((size_t)(b * HID + j)) * MODES] * invS;  // fr mean = DC/S
  __syncthreads();
  float hv = b1[j];
  for (int t = 0; t < 3 * HID; ++t) hv += g[t] * W1[t * 64 + j];
  h1[j] = fmaxf(hv, 0.f);
  __syncthreads();
  if (j < 3) {
    float lv = b2[j];
    for (int t = 0; t < 64; ++t) lv += h1[t] * W2[t * 3 + j];
    lg[j] = lv;
  }
  __syncthreads();
  if (j == 0) {
    float mx = fmaxf(lg[0], fmaxf(lg[1], lg[2]));
    float e0 = expf(lg[0] - mx), e1 = expf(lg[1] - mx), e2 = expf(lg[2] - mx);
    float s = e0 + e1 + e2;
    gatew[b * 3 + 0] = e0 / s; gatew[b * 3 + 1] = e1 / s; gatew[b * 3 + 2] = e2 / s;
  }
}

// ---------------------------------------------------------------------------
// Fused inverse transform (gate weights folded into coef matrix) + spatial
// gather + exact GELU, in place. One v_wmma_f32_16x16x32_f16 per 16x16 tile.
// ---------------------------------------------------------------------------
__global__ void eng_fused(const float* __restrict__ ofbuf, const float* __restrict__ frproj,
                          const int* __restrict__ idxb, const float* __restrict__ emb,
                          const float* __restrict__ spW, const float* __restrict__ spb,
                          const float* __restrict__ gatew, const _Float16* __restrict__ invB,
                          float* __restrict__ h) {
  int b = blockIdx.x >> 2, ct = blockIdx.x & 3;
  int tid = threadIdx.x, wv = tid >> 5, lane = tid & 31;
  float w0 = gatew[b * 3 + 0], w1 = gatew[b * 3 + 1], w2 = gatew[b * 3 + 2];
  const float invS = 1.0f / (float)S_LEN;

  // A operand (16x32 f16 coef matrix), built per-lane with contiguous k = 16*(lane/16) + el.
  int mrow = lane & 15;
  int cch = ct * 16 + mrow;
  const float* ofp = ofbuf + ((size_t)(b * HID + cch)) * 32;
  const float* prp = frproj + ((size_t)(b * HID + cch)) * MODES;
  float ccos[16], csin[16];
  for (int k = 0; k < 16; ++k) {
    float scale = (k == 0) ? invS : 2.0f * invS;
    ccos[k] = scale * (w1 * ofp[k] + w2 * prp[k]);
    csin[k] = (k == 0) ? 0.0f : -2.0f * invS * w1 * ofp[16 + k];
  }
  union { v16h v; _Float16 e[16]; } A;
  int hi8 = (lane < 16) ? 0 : 8;
  for (int el = 0; el < 16; ++el) {
    int v = el >> 1, hs = el & 1;
    int K = 2 * hi8 + el;
    A.e[el] = (_Float16)((K < 16) ? ccos[K] : csin[K - 16]);
  }

  // Hoist spatial-engram weight columns for this lane's 8 output rows.
  float wreg[8][ED], bbv[8]; int crow[8];
  for (int v = 0; v < 8; ++v) {
    int c = ct * 16 + v + hi8;
    crow[v] = c; bbv[v] = spb[c];
    for (int e = 0; e < ED; ++e) wreg[v][e] = spW[e * HID + c];
  }

  const int tbase = blockIdx.y * ((S_LEN / 16) / SSPLIT);
  const int tend = tbase + (S_LEN / 16) / SSPLIT;
  for (int t = tbase + wv; t < tend; t += 8) {
    if (t + 8 < tend) __builtin_prefetch(invB + (size_t)(t + 8) * 512 + lane * 16, 0, 0);
    v16h Bv = *reinterpret_cast<const v16h*>(invB + (size_t)t * 512 + lane * 16);
    v8f z = {};
    v8f d = __builtin_amdgcn_wmma_f32_16x16x32_f16(false, A.v, false, Bv, (short)0, z, false, false);
    int s = t * 16 + (lane & 15);
    int iv = idxb[(size_t)b * S_LEN + s];
    const float* e8 = emb + (size_t)iv * ED;
    float ev[ED];
    for (int e = 0; e < ED; ++e) ev[e] = e8[e];
    for (int v = 0; v < 8; ++v) {
      float spv = bbv[v];
      for (int e = 0; e < ED; ++e) spv += ev[e] * wreg[v][e];
      float val = d[v] + w0 * spv;
      float gl = 0.5f * val * (1.0f + erff(val * 0.70710678118654752f));
      h[((size_t)(b * HID + crow[v])) * S_LEN + s] = gl;
    }
  }
}

// Projection: out[b,0,s] = sum_d h[b,d,s]*W[d] + bias
__global__ void eng_proj(const float* __restrict__ h, const float* __restrict__ W,
                         const float* __restrict__ bias, float* __restrict__ out) {
  int b = blockIdx.x;
  int s = blockIdx.y * 256 + threadIdx.x;
  float acc = bias[0];
  for (int d = 0; d < HID; ++d) acc += h[((size_t)(b * HID + d)) * S_LEN + s] * W[d];
  out[(size_t)b * S_LEN + s] = acc;
}

// ---------------------------------------------------------------------------
extern "C" void kernel_launch(void* const* d_in, const int* in_sizes, int n_in,
                              void* d_out, int out_size, void* d_ws, size_t ws_size,
                              hipStream_t stream) {
  (void)in_sizes; (void)n_in; (void)out_size; (void)ws_size;
  const float* x      = (const float*)d_in[0];
  const float* liftW  = (const float*)d_in[1];
  const float* liftB  = (const float*)d_in[2];
  const float* projW  = (const float*)d_in[3];
  const float* projB  = (const float*)d_in[4];
  const float* sp_emb = (const float*)d_in[5];
  const float* sp_W   = (const float*)d_in[6];
  const float* sp_b   = (const float*)d_in[7];
  const float* fr_emb = (const float*)d_in[8];
  const float* fr_W   = (const float*)d_in[9];
  const float* fr_b   = (const float*)d_in[10];
  const float* g_W1   = (const float*)d_in[11];
  const float* g_b1   = (const float*)d_in[12];
  const float* g_W2   = (const float*)d_in[13];
  const float* g_b2   = (const float*)d_in[14];
  const float* mhfWr  = (const float*)d_in[15];
  const float* mhfWi  = (const float*)d_in[16];
  float* out = (float*)d_out;

  char* wsp = (char*)d_ws; size_t off = 0;
  auto carve = [&](size_t bytes) -> char* {
    char* p = wsp + off; off += (bytes + 255) & ~(size_t)255; return p;
  };
  float*    h       = (float*)carve(sizeof(float) * BATCH * HID * S_LEN);
  float*    sC      = (float*)carve(sizeof(float) * BATCH * S_LEN);
  float*    sCpart  = (float*)carve(sizeof(float) * 4 * BATCH * S_LEN);
  int*      idxb    = (int*)  carve(sizeof(int)   * BATCH * S_LEN);
  float*    modes   = (float*)carve(sizeof(float) * BATCH * HID * 32);
  float*    modp    = (float*)carve(sizeof(float) * KSPLIT * BATCH * HID * 32);
  float*    ofbuf   = (float*)carve(sizeof(float) * BATCH * HID * 32);
  float*    frproj  = (float*)carve(sizeof(float) * BATCH * HID * MODES);
  float*    embpart = (float*)carve(sizeof(float) * BATCH * SSPLIT * ED);
  float*    gatew   = (float*)carve(sizeof(float) * BATCH * 3);
  _Float16* invB    = (_Float16*)carve(sizeof(_Float16) * 32 * S_LEN);
  float*    fwdB    = (float*)carve(sizeof(float) * 32 * S_LEN);

  eng_init_invB<<<2048, 256, 0, stream>>>(invB);
  eng_init_fwdB<<<2048, 256, 0, stream>>>(fwdB);
  eng_lift<<<dim3(BATCH * HID, S_LEN / 256), 256, 0, stream>>>(x, liftW, liftB, h);

  for (int lay = 0; lay < LAYERS; ++lay) {
    eng_fwd_dft<<<dim3(BATCH * 4, KSPLIT), 128, 0, stream>>>(h, fwdB, modp, sCpart);
    eng_modes_red<<<(BATCH * HID * 32) / 256, 256, 0, stream>>>(modp, modes);
    eng_scomb<<<dim3(BATCH, S_LEN / 256), 256, 0, stream>>>(sCpart, sC);
    eng_spatial<<<dim3(BATCH, SSPLIT), 256, 0, stream>>>(sC, sp_emb + (size_t)lay * NP_SP * ED,
                                                         idxb, embpart);
    eng_mhf<<<64, 256, 0, stream>>>(modes, mhfWr + (size_t)lay * 4 * 16 * 16 * 16,
                                    mhfWi + (size_t)lay * 4 * 16 * 16 * 16, ofbuf);
    eng_freng<<<BATCH, 256, 0, stream>>>(modes, fr_emb + (size_t)lay * NP_FR * ED,
                                         fr_W + (size_t)lay * ED * HID * MODES,
                                         fr_b + (size_t)lay * HID * MODES, frproj);
    eng_gate<<<BATCH, 64, 0, stream>>>(embpart, ofbuf, frproj,
                                       sp_W + (size_t)lay * ED * HID, sp_b + (size_t)lay * HID,
                                       g_W1 + (size_t)lay * 3 * HID * 64, g_b1 + (size_t)lay * 64,
                                       g_W2 + (size_t)lay * 64 * 3, g_b2 + (size_t)lay * 3, gatew);
    eng_fused<<<dim3(BATCH * 4, SSPLIT), 256, 0, stream>>>(ofbuf, frproj, idxb,
                                                           sp_emb + (size_t)lay * NP_SP * ED,
                                                           sp_W + (size_t)lay * ED * HID,
                                                           sp_b + (size_t)lay * HID, gatew, invB, h);
  }
  eng_proj<<<dim3(BATCH, S_LEN / 256), 256, 0, stream>>>(h, projW, projB, out);
}
